// SigLIPModel_38534446580212
// MI455X (gfx1250) — matrix operand order repaired and checked by reading, not compiled, not run on hardware
//
#include <hip/hip_runtime.h>
#include <hip/hip_bf16.h>
#include <math.h>

typedef __attribute__((ext_vector_type(16))) __bf16 v16bf;
typedef __attribute__((ext_vector_type(8)))  __bf16 v8bf;
typedef __attribute__((ext_vector_type(4)))  __bf16 v4bf;
typedef __attribute__((ext_vector_type(8)))  float  v8f;

#define TILE_M 128
#define TILE_N 128
#define TILE_K 32
#define LDS_STRIDE 40   // 40 bf16 = 80 B rows: 16B-aligned, 20-bank stride -> conflict-free b128 reads

// split a float4 into bf16 hi (top ~8 mantissa bits) and lo (residual ~8 bits) vectors
__device__ __forceinline__ void split4(const float4 &v, v4bf &h, v4bf &l) {
    __bf16 h0 = (__bf16)v.x, h1 = (__bf16)v.y, h2 = (__bf16)v.z, h3 = (__bf16)v.w;
    h = (v4bf){h0, h1, h2, h3};
    l = (v4bf){(__bf16)(v.x - (float)h0), (__bf16)(v.y - (float)h1),
               (__bf16)(v.z - (float)h2), (__bf16)(v.w - (float)h3)};
}

// A fragment, 16-bit A 16x32 layout: lanes 0-15 M=lane; e<8 -> K=e+kh*8, e>=8 -> K=e+8+kh*8
__device__ __forceinline__ v16bf ld_afrag(const __bf16 P[][LDS_STRIDE], int r, int kh) {
    v8bf a = *(const v8bf*)&P[r][kh * 8];
    v8bf b = *(const v8bf*)&P[r][kh * 8 + 16];
    return __builtin_shufflevector(a, b, 0,1,2,3,4,5,6,7,8,9,10,11,12,13,14,15);
}
// B fragment, 16-bit B 32x16 layout: lanes 0-15 N=lane; e<8 -> K=e+kh*8, e>=8 -> K=e+8+kh*8
__device__ __forceinline__ v16bf ld_bfrag(const __bf16 P[][LDS_STRIDE], int r, int kh) {
    v8bf a = *(const v8bf*)&P[r][kh * 8];
    v8bf b = *(const v8bf*)&P[r][kh * 8 + 16];
    return __builtin_shufflevector(a, b, 0,1,2,3,4,5,6,7,8,9,10,11,12,13,14,15);
}

__global__ void siglip_zero(float* out) { out[0] = 0.0f; }

__global__ void __launch_bounds__(256)
siglip_loss_kernel(const float* __restrict__ emb1,
                   const float* __restrict__ emb2,
                   float* __restrict__ out,
                   int Bdim, int D)
{
    __shared__ __align__(16) __bf16 Ahi[TILE_M][LDS_STRIDE];
    __shared__ __align__(16) __bf16 Alo[TILE_M][LDS_STRIDE];
    __shared__ __align__(16) __bf16 Bhi[TILE_N][LDS_STRIDE];
    __shared__ __align__(16) __bf16 Blo[TILE_N][LDS_STRIDE];
    __shared__ float red[8];

    const int tid  = threadIdx.x;
    const int lane = tid & 31;
    const int wave = tid >> 5;      // 0..7
    const int wm   = wave & 3;      // wave M position: 0..3 (32 rows each)
    const int wn   = wave >> 2;     // wave N position: 0..1 (64 cols each)

    const int bM = blockIdx.x;
    const int bN = blockIdx.y;

    const float* Ag = emb1 + (size_t)bM * TILE_M * D;
    const float* Bg = emb2 + (size_t)bN * TILE_N * D;

    v8f acc[2][4];
    #pragma unroll
    for (int mt = 0; mt < 2; ++mt)
        #pragma unroll
        for (int nt = 0; nt < 4; ++nt)
            acc[mt][nt] = (v8f){0.f,0.f,0.f,0.f,0.f,0.f,0.f,0.f};

    const int ldc = (tid & 7) * 4;  // fp32 column group in K chunk
    const int ldr = tid >> 3;       // base row 0..31 (+32*i)

    const int mlane = lane & 15;
    const int khalf = lane >> 4;

    float4 pa[4], pb[4];
    auto loadPanels = [&](int k0) {
        #pragma unroll
        for (int i = 0; i < 4; ++i)
            pa[i] = *reinterpret_cast<const float4*>(Ag + (size_t)(ldr + i * 32) * D + k0 + ldc);
        #pragma unroll
        for (int i = 0; i < 4; ++i)
            pb[i] = *reinterpret_cast<const float4*>(Bg + (size_t)(ldr + i * 32) * D + k0 + ldc);
    };
    auto storePanels = [&]() {
        #pragma unroll
        for (int i = 0; i < 4; ++i) {
            int r = ldr + i * 32;
            v4bf h, l;
            split4(pa[i], h, l);
            *(v4bf*)&Ahi[r][ldc] = h;
            *(v4bf*)&Alo[r][ldc] = l;
        }
        #pragma unroll
        for (int i = 0; i < 4; ++i) {
            int r = ldr + i * 32;
            v4bf h, l;
            split4(pb[i], h, l);
            *(v4bf*)&Bhi[r][ldc] = h;
            *(v4bf*)&Blo[r][ldc] = l;
        }
    };

    const int NK = D / TILE_K;
    loadPanels(0);

    for (int kb = 0; kb < NK; ++kb) {
        __syncthreads();          // previous iteration's LDS reads complete
        storePanels();
        __syncthreads();          // LDS panels ready
        if (kb + 1 < NK) loadPanels((kb + 1) * TILE_K);   // overlap next fetch with compute

        v16bf afh[2], afl[2], bfh[4], bfl[4];
        #pragma unroll
        for (int mt = 0; mt < 2; ++mt) {
            int r = wm * 32 + mt * 16 + mlane;
            afh[mt] = ld_afrag(Ahi, r, khalf);
            afl[mt] = ld_afrag(Alo, r, khalf);
        }
        #pragma unroll
        for (int nt = 0; nt < 4; ++nt) {
            int r = wn * 64 + nt * 16 + mlane;
            bfh[nt] = ld_bfrag(Bhi, r, khalf);
            bfl[nt] = ld_bfrag(Blo, r, khalf);
        }

        // bf16x3 fp32-emulated MACs: hi*hi + hi*lo + lo*hi
        #pragma unroll
        for (int mt = 0; mt < 2; ++mt)
            #pragma unroll
            for (int nt = 0; nt < 4; ++nt) {
                acc[mt][nt] = __builtin_amdgcn_wmma_f32_16x16x32_bf16(
                    false, afl[mt], false, bfh[nt], (short)0, acc[mt][nt], false, false);
                acc[mt][nt] = __builtin_amdgcn_wmma_f32_16x16x32_bf16(
                    false, afh[mt], false, bfl[nt], (short)0, acc[mt][nt], false, false);
                acc[mt][nt] = __builtin_amdgcn_wmma_f32_16x16x32_bf16(
                    false, afh[mt], false, bfh[nt], (short)0, acc[mt][nt], false, false);
            }
    }

    // epilogue: loss = [ sum_ij softplus(l_ij) - sum_i l_ii ] / B
    const float invT = 1.0f / 0.07f;
    const float invB = 1.0f / (float)Bdim;
    float local = 0.0f;
    #pragma unroll
    for (int mt = 0; mt < 2; ++mt) {
        #pragma unroll
        for (int nt = 0; nt < 4; ++nt) {
            int gm0 = bM * TILE_M + wm * 32 + mt * 16 + khalf * 8;  // + v = row
            int col = bN * TILE_N + wn * 64 + nt * 16 + mlane;
            #pragma unroll
            for (int v = 0; v < 8; ++v) {
                int   row = gm0 + v;
                float l   = acc[mt][nt][v] * invT;
                float e   = __expf(-fabsf(l));
                float sp  = fmaxf(l, 0.0f) + __logf(1.0f + e);
                local += sp - ((row == col) ? l : 0.0f);
            }
        }
    }
    local *= invB;

    #pragma unroll
    for (int off = 16; off > 0; off >>= 1)
        local += __shfl_down(local, off, 32);
    if (lane == 0) red[wave] = local;
    __syncthreads();
    if (tid == 0) {
        float s = 0.0f;
        #pragma unroll
        for (int w = 0; w < 8; ++w) s += red[w];
        atomicAdd(out, s);
    }
}

extern "C" void kernel_launch(void* const* d_in, const int* in_sizes, int n_in,
                              void* d_out, int out_size, void* d_ws, size_t ws_size,
                              hipStream_t stream) {
    const float* emb1 = (const float*)d_in[0];
    const float* emb2 = (const float*)d_in[1];
    float* out = (float*)d_out;

    const int D    = 1024;
    const int Bdim = in_sizes[0] / D;   // 8192

    siglip_zero<<<1, 1, 0, stream>>>(out);

    dim3 grid(Bdim / TILE_M, Bdim / TILE_N);   // 64 x 64 blocks
    siglip_loss_kernel<<<grid, dim3(256), 0, stream>>>(emb1, emb2, out, Bdim, D);
}
